// EvolveGCN_87892210746082
// MI455X (gfx1250) — matrix operand order repaired and checked by reading, not compiled, not run on hardware
//
#include <hip/hip_runtime.h>

// ---------------------------------------------------------------------------
// EvolveGCN-O for MI455X (gfx1250): bf16 WMMA GEMMs + atomic edge aggregation
// v2: transposed-W LDS staging (b128 fragment loads), fused cvt/RReLU in GEMM
// ---------------------------------------------------------------------------

#define RRELU_SLOPE 0.2291666716337204f  // (1/8 + 1/3) / 2

typedef __attribute__((ext_vector_type(16))) __bf16 v16bf;
typedef __attribute__((ext_vector_type(8)))  float  v8f;

union Frag16 { v16bf v; unsigned short u[16]; uint4 q[2]; };

__device__ __forceinline__ unsigned short f32_to_bf16_rne(float f) {
  unsigned u = __float_as_uint(f);
  u += 0x7FFFu + ((u >> 16) & 1u);   // round to nearest even
  return (unsigned short)(u >> 16);
}

// ---------------------------------------------------------------- utilities
__global__ void zero_f32(float* __restrict__ p, int n) {
  int i = blockIdx.x * blockDim.x + threadIdx.x;
  int stride = gridDim.x * blockDim.x;
  for (; i < n; i += stride) p[i] = 0.0f;
}

// W [128][128] f32 row-major -> bf16 TRANSPOSED [n][k] (for B-fragment loads)
__global__ void cvt_w_bf16_t(const float* __restrict__ x,
                             unsigned short* __restrict__ y) {
  int idx = blockIdx.x * blockDim.x + threadIdx.x;
  if (idx >= 128 * 128) return;
  int i = idx >> 7, j = idx & 127;      // i = K row, j = N col of W
  y[j * 128 + i] = f32_to_bf16_rne(x[idx]);
}

// copy [128 x cols] row-major into [128 x 128] padded (pad cols = 0)
__global__ void copy_pad(const float* __restrict__ src, float* __restrict__ dst,
                         int cols) {
  int idx = blockIdx.x * blockDim.x + threadIdx.x;
  if (idx >= 128 * 128) return;
  int i = idx >> 7, j = idx & 127;
  dst[idx] = (j < cols) ? src[i * cols + j] : 0.0f;
}

// ------------------------------------------------------------- graph degrees
__global__ void degree_kernel(const int* __restrict__ src,
                              const int* __restrict__ dst,
                              float* __restrict__ degO,
                              float* __restrict__ degI, int E) {
  int e = blockIdx.x * blockDim.x + threadIdx.x;
  if (e < E) {
    atomicAdd(&degO[src[e]], 1.0f);
    atomicAdd(&degI[dst[e]], 1.0f);
  }
}

__global__ void norm_kernel(const int* __restrict__ src,
                            const int* __restrict__ dst,
                            const float* __restrict__ degO,
                            const float* __restrict__ degI,
                            float* __restrict__ norm, int E) {
  int e = blockIdx.x * blockDim.x + threadIdx.x;
  if (e < E) {
    float a = __frsqrt_rn(fmaxf(degO[src[e]], 1.0f));
    float b = __frsqrt_rn(fmaxf(degI[dst[e]], 1.0f));
    norm[e] = a * b;
  }
}

// ------------------------------------------------------------------ mat-GRU
__global__ void gru_phase1(const float* __restrict__ gW,
                           const float* __restrict__ gU,
                           const float* __restrict__ b,
                           const float* __restrict__ Wh,
                           float* __restrict__ zbuf,
                           float* __restrict__ rWh, int Ccols) {
  int idx = blockIdx.x * blockDim.x + threadIdx.x;
  if (idx >= 128 * 128) return;
  int i = idx >> 7, j = idx & 127;
  if (j >= Ccols) { zbuf[idx] = 0.0f; rWh[idx] = 0.0f; return; }
  float sZ = b[(0 * 128 + i) * Ccols + j];
  float sR = b[(1 * 128 + i) * Ccols + j];
  const float* w0 = gW + (0 * 128 + i) * 128;
  const float* u0 = gU + (0 * 128 + i) * 128;
  const float* w1 = gW + (1 * 128 + i) * 128;
  const float* u1 = gU + (1 * 128 + i) * 128;
  for (int k = 0; k < 128; ++k) {
    float whkj = Wh[k * 128 + j];
    sZ = fmaf(w0[k] + u0[k], whkj, sZ);
    sR = fmaf(w1[k] + u1[k], whkj, sR);
  }
  float z = 1.0f / (1.0f + __expf(-sZ));
  float r = 1.0f / (1.0f + __expf(-sR));
  zbuf[idx] = z;
  rWh[idx]  = r * Wh[idx];
}

__global__ void gru_phase2(const float* __restrict__ gW,
                           const float* __restrict__ gU,
                           const float* __restrict__ b,
                           const float* __restrict__ Wh,
                           const float* __restrict__ zbuf,
                           const float* __restrict__ rWh,
                           float* __restrict__ Wout, int Ccols) {
  int idx = blockIdx.x * blockDim.x + threadIdx.x;
  if (idx >= 128 * 128) return;
  int i = idx >> 7, j = idx & 127;
  if (j >= Ccols) { Wout[idx] = 0.0f; return; }
  float sH = b[(2 * 128 + i) * Ccols + j];
  const float* w2 = gW + (2 * 128 + i) * 128;
  const float* u2 = gU + (2 * 128 + i) * 128;
  for (int k = 0; k < 128; ++k) {
    sH = fmaf(w2[k], Wh[k * 128 + j], sH);
    sH = fmaf(u2[k], rWh[k * 128 + j], sH);
  }
  float ht = tanhf(sH);
  float z  = zbuf[idx];
  Wout[idx] = (1.0f - z) * Wh[idx] + z * ht;
}

// --------------------------------------------------------------- WMMA GEMM
// Y[n,128] (f32) = leaky?(X[n,128] f32) @ W (bf16, given TRANSPOSED [n][k]).
// Block = 256 threads = 8 waves; block tile = 128 rows; wave tile = 16 rows.
// f32 -> bf16 conversion (and optional RReLU) fused into LDS staging.
// Fragments load as contiguous b128 from LDS (A: row-major X; B: W^T rows).
__global__ __launch_bounds__(256) void gemm_bf16_wmma(
    const float* __restrict__ X,
    const unsigned short* __restrict__ Wt,   // [128][128] bf16, W transposed
    float* __restrict__ Y, int n, int leaky) {
  __shared__ unsigned short sX[128 * 128];   // 32 KB, row-major [m][k]
  __shared__ unsigned short sWt[128 * 128];  // 32 KB, [ncol][k]
  const int tid = threadIdx.x;
  const int rowBase = blockIdx.x * 128;

  // stage X: float4 global loads -> bf16x4 packed -> ds_store_b64
  for (int idx = tid; idx < 128 * 32; idx += 256) {
    int rr = idx >> 5;              // row in tile
    int c4 = (idx & 31);            // which float4 in the row
    int grow = rowBase + rr;
    float4 v = make_float4(0.f, 0.f, 0.f, 0.f);
    if (grow < n) v = ((const float4*)(X + (size_t)grow * 128))[c4];
    if (leaky) {
      v.x = (v.x >= 0.f) ? v.x : RRELU_SLOPE * v.x;
      v.y = (v.y >= 0.f) ? v.y : RRELU_SLOPE * v.y;
      v.z = (v.z >= 0.f) ? v.z : RRELU_SLOPE * v.z;
      v.w = (v.w >= 0.f) ? v.w : RRELU_SLOPE * v.w;
    }
    union { unsigned short s[4]; uint2 u; } pk;
    pk.s[0] = f32_to_bf16_rne(v.x);
    pk.s[1] = f32_to_bf16_rne(v.y);
    pk.s[2] = f32_to_bf16_rne(v.z);
    pk.s[3] = f32_to_bf16_rne(v.w);
    *(uint2*)&sX[rr * 128 + c4 * 4] = pk.u;
  }
  // stage W^T: straight vector copy (2048 uint4)
  for (int idx = tid; idx < 2048; idx += 256) {
    ((uint4*)sWt)[idx] = ((const uint4*)Wt)[idx];
  }
  __syncthreads();

  const int wave = tid >> 5;
  const int lane = tid & 31;
  const int hi   = lane >> 4;       // lane half
  const int lp   = lane & 15;
  const int mlds = wave * 16 + lp;  // A-matrix row for this lane

  v8f acc[8];
#pragma unroll
  for (int nt = 0; nt < 8; ++nt) {
    v8f z = {0.f, 0.f, 0.f, 0.f, 0.f, 0.f, 0.f, 0.f};
    acc[nt] = z;
  }

#pragma unroll
  for (int kc = 0; kc < 128; kc += 32) {
    // A fragment (16-bit A 16x32, ISA 7.12.2): two contiguous 8-elem runs
    Frag16 a;
    a.q[0] = *(const uint4*)&sX[mlds * 128 + kc + 8 * hi];        // K 0..7 (+8h)
    a.q[1] = *(const uint4*)&sX[mlds * 128 + kc + 16 + 8 * hi];   // K 16..23 (+8h)
#pragma unroll
    for (int nt = 0; nt < 8; ++nt) {
      // B fragment: same K order as A (K 0..7 / 16..23, +8h), from W^T row
      Frag16 bfr;
      int ncol = nt * 16 + lp;
      bfr.q[0] = *(const uint4*)&sWt[ncol * 128 + kc + 8 * hi];
      bfr.q[1] = *(const uint4*)&sWt[ncol * 128 + kc + 16 + 8 * hi];
      acc[nt] = __builtin_amdgcn_wmma_f32_16x16x32_bf16(
          false, a.v, false, bfr.v, (short)0, acc[nt], false, false);
    }
  }

#pragma unroll
  for (int nt = 0; nt < 8; ++nt) {
    const float* af = (const float*)&acc[nt];
#pragma unroll
    for (int v = 0; v < 8; ++v) {
      int row = rowBase + wave * 16 + v + 8 * hi;  // D: M = vgpr + 8*half
      if (row < n) Y[(size_t)row * 128 + nt * 16 + lp] = af[v];
    }
  }
}

// ---------------------------------------------------- edge scatter-aggregate
__global__ void aggregate_kernel(const float* __restrict__ X,
                                 const int* __restrict__ src,
                                 const int* __restrict__ dst,
                                 const float* __restrict__ norm,
                                 float* __restrict__ out, int E) {
  long long gid = (long long)blockIdx.x * blockDim.x + threadIdx.x;
  int e = (int)(gid >> 5);
  if (e >= E) return;
  int lane = (int)(gid & 31);
  int s = src[e], d = dst[e];
  float w = norm[e];
  float4 v = ((const float4*)(X + (size_t)s * 128))[lane];
  float* o = out + (size_t)d * 128 + lane * 4;
  atomicAdd(o + 0, v.x * w);
  atomicAdd(o + 1, v.y * w);
  atomicAdd(o + 2, v.z * w);
  atomicAdd(o + 3, v.w * w);
}

// ------------------------------------------------------------------ MLP head
__global__ __launch_bounds__(128) void mlp_kernel(
    const float* __restrict__ h, const float* __restrict__ w1,
    const float* __restrict__ b1, const float* __restrict__ w2,
    const float* __restrict__ b2, float* __restrict__ out, int n) {
  __shared__ float sw1[127 * 64];  // 31.75 KB
  __shared__ float sw2[64 * 2];
  __shared__ float sb1[64];
  for (int i = threadIdx.x; i < 127 * 64; i += blockDim.x) sw1[i] = w1[i];
  for (int i = threadIdx.x; i < 128; i += blockDim.x) sw2[i] = w2[i];
  for (int i = threadIdx.x; i < 64; i += blockDim.x) sb1[i] = b1[i];
  __syncthreads();
  int node = blockIdx.x * blockDim.x + threadIdx.x;
  if (node >= n) return;
  float hid[64];
#pragma unroll
  for (int c = 0; c < 64; ++c) hid[c] = sb1[c];
  const float* row = h + (size_t)node * 128;  // padded, col 127 == 0
  for (int k = 0; k < 127; ++k) {
    float x = row[k];
#pragma unroll
    for (int c = 0; c < 64; ++c) hid[c] = fmaf(x, sw1[k * 64 + c], hid[c]);
  }
  float o0 = b2[0], o1 = b2[1];
#pragma unroll
  for (int c = 0; c < 64; ++c) {
    float a = fmaxf(hid[c], 0.0f);
    o0 = fmaf(a, sw2[c * 2 + 0], o0);
    o1 = fmaf(a, sw2[c * 2 + 1], o1);
  }
  out[(size_t)node * 2 + 0] = o0;
  out[(size_t)node * 2 + 1] = o1;
}

// ---------------------------------------------------------------------------
extern "C" void kernel_launch(void* const* d_in, const int* in_sizes, int n_in,
                              void* d_out, int out_size, void* d_ws,
                              size_t ws_size, hipStream_t stream) {
  const float* feats = (const float*)d_in[0];
  const int*   src   = (const int*)d_in[1];
  const int*   dst   = (const int*)d_in[2];
  const float* W1_0  = (const float*)d_in[3];
  const float* W2_0  = (const float*)d_in[4];
  const float* g1W   = (const float*)d_in[5];
  const float* g1U   = (const float*)d_in[6];
  const float* g1b   = (const float*)d_in[7];
  const float* g2W   = (const float*)d_in[8];
  const float* g2U   = (const float*)d_in[9];
  const float* g2b   = (const float*)d_in[10];
  const float* mw1   = (const float*)d_in[11];
  const float* mb1   = (const float*)d_in[12];
  const float* mw2   = (const float*)d_in[13];
  const float* mb2   = (const float*)d_in[14];
  float* out = (float*)d_out;

  const int T = 8;
  const int N = in_sizes[0] / (T * 128);
  const int E = in_sizes[1];
  const int ND = N * 128;

  // workspace carve-out
  char* p = (char*)d_ws;
  auto alloc = [&](size_t bytes) {
    void* r = (void*)p;
    p += (bytes + 255) & ~(size_t)255;
    return r;
  };
  float* degO = (float*)alloc((size_t)N * 4);
  float* degI = (float*)alloc((size_t)N * 4);
  float* norm = (float*)alloc((size_t)E * 4);
  float* W1a  = (float*)alloc(128 * 128 * 4);
  float* W1b  = (float*)alloc(128 * 128 * 4);
  float* W2a  = (float*)alloc(128 * 128 * 4);
  float* W2b  = (float*)alloc(128 * 128 * 4);
  float* zbuf = (float*)alloc(128 * 128 * 4);
  float* rWh  = (float*)alloc(128 * 128 * 4);
  unsigned short* W1bf = (unsigned short*)alloc(128 * 128 * 2);
  unsigned short* W2bf = (unsigned short*)alloc(128 * 128 * 2);
  float* gout = (float*)alloc((size_t)ND * 4);
  float* h1   = (float*)alloc((size_t)ND * 4);
  float* h2   = (float*)alloc((size_t)ND * 4);

  const int eb = (E + 255) / 256;
  const int ab = (int)(((long long)E * 32 + 255) / 256);
  const int gb = (N + 127) / 128;
  const int mb = (N + 127) / 128;

  // graph normalization (once)
  zero_f32<<<256, 256, 0, stream>>>(degO, N);
  zero_f32<<<256, 256, 0, stream>>>(degI, N);
  degree_kernel<<<eb, 256, 0, stream>>>(src, dst, degO, degI, E);
  norm_kernel<<<eb, 256, 0, stream>>>(src, dst, degO, degI, norm, E);

  // initial GCN weights (padded to 128 cols)
  copy_pad<<<64, 256, 0, stream>>>(W1_0, W1a, 128);
  copy_pad<<<64, 256, 0, stream>>>(W2_0, W2a, 127);

  float* W1cur = W1a; float* W1alt = W1b;
  float* W2cur = W2a; float* W2alt = W2b;

  for (int t = 0; t < T; ++t) {
    // evolve W1 = matGRU(W1), W2 = matGRU(W2)
    gru_phase1<<<64, 256, 0, stream>>>(g1W, g1U, g1b, W1cur, zbuf, rWh, 128);
    gru_phase2<<<64, 256, 0, stream>>>(g1W, g1U, g1b, W1cur, zbuf, rWh, W1alt, 128);
    { float* tmp = W1cur; W1cur = W1alt; W1alt = tmp; }
    gru_phase1<<<64, 256, 0, stream>>>(g2W, g2U, g2b, W2cur, zbuf, rWh, 127);
    gru_phase2<<<64, 256, 0, stream>>>(g2W, g2U, g2b, W2cur, zbuf, rWh, W2alt, 127);
    { float* tmp = W2cur; W2cur = W2alt; W2alt = tmp; }

    // bf16 transposed operands for WMMA B-matrix
    cvt_w_bf16_t<<<64, 256, 0, stream>>>(W1cur, W1bf);
    cvt_w_bf16_t<<<64, 256, 0, stream>>>(W2cur, W2bf);

    // h1 = aggregate(x @ W1)   (cvt fused into gemm staging)
    gemm_bf16_wmma<<<gb, 256, 0, stream>>>(feats + (size_t)t * ND, W1bf, gout,
                                           N, 0);
    zero_f32<<<2048, 256, 0, stream>>>(h1, ND);
    aggregate_kernel<<<ab, 256, 0, stream>>>(gout, src, dst, norm, h1, E);

    // h2 = aggregate(rrelu(h1) @ W2)  (RReLU fused into gemm staging)
    gemm_bf16_wmma<<<gb, 256, 0, stream>>>(h1, W2bf, gout, N, 1);
    zero_f32<<<2048, 256, 0, stream>>>(h2, ND);
    aggregate_kernel<<<ab, 256, 0, stream>>>(gout, src, dst, norm, h2, E);
  }

  // MLP head
  mlp_kernel<<<mb, 128, 0, stream>>>(h2, mw1, mb1, mw2, mb2, out, N);
}
